// DDiTBlock_88416196755847
// MI455X (gfx1250) — matrix operand order repaired and checked by reading, compiled, not run on hardware
//
#include <hip/hip_runtime.h>
#include <hip/hip_bf16.h>

// ---------------------------------------------------------------------------
// DiT block for MI455X (gfx1250): bf16 WMMA (16x16x32) GEMMs + flash attention
// Wave tile = 32x64 (2 A-frags x 4 B-frags = 8 wmma per k-step) to raise
// arithmetic intensity; global_prefetch for streaming operands.
// ---------------------------------------------------------------------------

#define T_SEQ 4096
#define DIM   1024
#define HEADS 16
#define HDIM  64
#define MLPD  4096
#define ADA6  6144

typedef __bf16 v16bf __attribute__((ext_vector_type(16)));
typedef float  v8f   __attribute__((ext_vector_type(8)));

union Frag {
    v16bf v;
    uint4 q[2];
};

__device__ __forceinline__ v8f wmma_bf16(const Frag& a, const Frag& b, v8f c) {
    return __builtin_amdgcn_wmma_f32_16x16x32_bf16(
        /*neg_a=*/false, a.v, /*neg_b=*/false, b.v,
        /*c_mod=*/(short)0, c, /*reuse_a=*/false, /*reuse_b=*/false);
}

__device__ __forceinline__ float gelu_tanh(float x) {
    float x3 = x * x * x;
    return 0.5f * x * (1.0f + tanhf(0.7978845608028654f * (x + 0.044715f * x3)));
}

// ---------------------------------------------------------------------------
// ada = c @ w_ada + b_ada   (1x1024 @ 1024x6144)
// ---------------------------------------------------------------------------
__global__ __launch_bounds__(256) void ada_kernel(
    const float* __restrict__ c, const float* __restrict__ w,
    const float* __restrict__ b, float* __restrict__ ada) {
    int j = blockIdx.x * 256 + threadIdx.x;
    float s = b[j];
    for (int k = 0; k < DIM; ++k)
        s = fmaf(c[k], w[(size_t)k * ADA6 + j], s);
    ada[j] = s;
}

// ---------------------------------------------------------------------------
// out[n*K + k] = (bf16) in[k*N + n]   : f32 KxN  ->  bf16 NxK (transposed)
// ---------------------------------------------------------------------------
__global__ __launch_bounds__(256) void convT_kernel(
    const float* __restrict__ in, __bf16* __restrict__ out, int K, int N) {
    size_t idx = (size_t)blockIdx.x * 256 + threadIdx.x;
    if (idx >= (size_t)K * N) return;
    size_t n = idx / K, k = idx % K;
    out[idx] = (__bf16)in[k * (size_t)N + n];
}

// ---------------------------------------------------------------------------
// h = (LN(x) * w + b) * (1 + sc) + sh     -> bf16, one block per row
// ---------------------------------------------------------------------------
__global__ __launch_bounds__(256) void lnmod_kernel(
    const float* __restrict__ x, const float* __restrict__ w,
    const float* __restrict__ b, const float* __restrict__ sc,
    const float* __restrict__ sh, __bf16* __restrict__ out) {
    int t = blockIdx.x, tid = threadIdx.x;
    const float* xr = x + (size_t)t * DIM;
    float v[4], s = 0.f;
#pragma unroll
    for (int i = 0; i < 4; ++i) { v[i] = xr[tid + 256 * i]; s += v[i]; }
    __shared__ float red[256];
    red[tid] = s; __syncthreads();
    for (int off = 128; off > 0; off >>= 1) {
        if (tid < off) red[tid] += red[tid + off];
        __syncthreads();
    }
    float mu = red[0] * (1.f / DIM);
    __syncthreads();
    float s2 = 0.f;
#pragma unroll
    for (int i = 0; i < 4; ++i) { float d = v[i] - mu; s2 += d * d; }
    red[tid] = s2; __syncthreads();
    for (int off = 128; off > 0; off >>= 1) {
        if (tid < off) red[tid] += red[tid + off];
        __syncthreads();
    }
    float rs = rsqrtf(red[0] * (1.f / DIM) + 1e-5f);
#pragma unroll
    for (int i = 0; i < 4; ++i) {
        int col = tid + 256 * i;
        float h = (v[i] - mu) * rs * w[col] + b[col];
        h = h * (1.f + sc[col]) + sh[col];
        out[(size_t)t * DIM + col] = (__bf16)h;
    }
}

// ---------------------------------------------------------------------------
// RoPE + split qkv (f32 [T][3*DIM]) -> q bf16 [T][DIM], k bf16 [T][DIM],
// vT bf16 [DIM][T]   (V transposed so the PV matmul B-fragments are contiguous)
// ---------------------------------------------------------------------------
__global__ __launch_bounds__(512) void rope_split_kernel(
    const float* __restrict__ qkv, const float* __restrict__ cosp,
    const float* __restrict__ sinp, __bf16* __restrict__ qb,
    __bf16* __restrict__ kb, __bf16* __restrict__ vT) {
    int t = blockIdx.x, tid = threadIdx.x;
    int h = tid >> 5, d = tid & 31;
    const float* base = qkv + (size_t)t * (3 * DIM);
    float cv = cosp[t * 32 + d], sv = sinp[t * 32 + d];
    int hd = h * HDIM + d;
    float q1 = base[hd], q2 = base[hd + 32];
    qb[(size_t)t * DIM + hd]      = (__bf16)(q1 * cv - q2 * sv);
    qb[(size_t)t * DIM + hd + 32] = (__bf16)(q2 * cv + q1 * sv);
    float k1 = base[DIM + hd], k2 = base[DIM + hd + 32];
    kb[(size_t)t * DIM + hd]      = (__bf16)(k1 * cv - k2 * sv);
    kb[(size_t)t * DIM + hd + 32] = (__bf16)(k2 * cv + k1 * sv);
    float v1 = base[2 * DIM + hd], v2 = base[2 * DIM + hd + 32];
    vT[(size_t)hd * T_SEQ + t]        = (__bf16)v1;
    vT[(size_t)(hd + 32) * T_SEQ + t] = (__bf16)v2;
}

// ---------------------------------------------------------------------------
// Flash attention: one wave per (32 query rows, head). Online softmax in f32.
// K/V tile fragments are loaded once per 32-key chunk and reused by both
// 16-row Q/P fragments (halves K/V traffic vs 16-row waves).
// ---------------------------------------------------------------------------
__global__ __launch_bounds__(32) void attn_kernel(
    const __bf16* __restrict__ qb, const __bf16* __restrict__ kb,
    const __bf16* __restrict__ vT, const int* __restrict__ seq,
    __bf16* __restrict__ ob) {
    const int lane = threadIdx.x & 31;
    const int col16 = lane & 15, hi = lane >> 4;
    const int t0 = blockIdx.x * 32;
    const int h = blockIdx.y;

    // Q fragments: [mt][k-half]
    Frag aq[2][2];
#pragma unroll
    for (int mt = 0; mt < 2; ++mt) {
        const __bf16* qrow = qb + (size_t)(t0 + mt * 16 + col16) * DIM + h * HDIM;
        aq[mt][0].q[0] = *(const uint4*)(qrow + hi * 8);
        aq[mt][0].q[1] = *(const uint4*)(qrow + 16 + hi * 8);
        aq[mt][1].q[0] = *(const uint4*)(qrow + 32 + hi * 8);
        aq[mt][1].q[1] = *(const uint4*)(qrow + 48 + hi * 8);
    }

    int srow[2][8];
#pragma unroll
    for (int mt = 0; mt < 2; ++mt)
#pragma unroll
        for (int r = 0; r < 8; ++r) srow[mt][r] = seq[t0 + mt * 16 + r + 8 * hi];

    float m[2][8], l[2][8];
#pragma unroll
    for (int mt = 0; mt < 2; ++mt)
#pragma unroll
        for (int r = 0; r < 8; ++r) { m[mt][r] = -1e30f; l[mt][r] = 0.f; }
    v8f acc[2][4] = {};

    __shared__ __align__(16) __bf16 pbuf[2][16][32];

    for (int s0 = 0; s0 < T_SEQ; s0 += 32) {
        if (s0 + 32 < T_SEQ) {
            __builtin_prefetch(kb + (size_t)(s0 + 32 + col16) * DIM + h * HDIM, 0, 3);
            __builtin_prefetch(vT + (size_t)(h * HDIM + col16) * T_SEQ + s0 + 32, 0, 3);
        }
        // K tile fragments, shared by both Q row-tiles: [j][k-half]
        Frag bk[2][2];
#pragma unroll
        for (int j = 0; j < 2; ++j) {
            const __bf16* krow = kb + (size_t)(s0 + j * 16 + col16) * DIM + h * HDIM;
            bk[j][0].q[0] = *(const uint4*)(krow + hi * 8);
            bk[j][0].q[1] = *(const uint4*)(krow + 16 + hi * 8);
            bk[j][1].q[0] = *(const uint4*)(krow + 32 + hi * 8);
            bk[j][1].q[1] = *(const uint4*)(krow + 48 + hi * 8);
        }
        int sc0 = seq[s0 + col16];
        int sc1 = seq[s0 + 16 + col16];

        __syncthreads();   // single-wave WG: hardware NOP, compiler ordering
#pragma unroll
        for (int mt = 0; mt < 2; ++mt) {
            v8f S0 = {}, S1 = {};
            S0 = wmma_bf16(aq[mt][0], bk[0][0], S0);
            S0 = wmma_bf16(aq[mt][1], bk[0][1], S0);
            S1 = wmma_bf16(aq[mt][0], bk[1][0], S1);
            S1 = wmma_bf16(aq[mt][1], bk[1][1], S1);
            float p0[8], p1[8];
#pragma unroll
            for (int r = 0; r < 8; ++r) {
                float v0 = (srow[mt][r] == sc0) ? S0[r] * 0.125f : -1e30f;
                float v1 = (srow[mt][r] == sc1) ? S1[r] * 0.125f : -1e30f;
                float mx = fmaxf(v0, v1);
                mx = fmaxf(mx, __shfl_xor(mx, 1, 32));
                mx = fmaxf(mx, __shfl_xor(mx, 2, 32));
                mx = fmaxf(mx, __shfl_xor(mx, 4, 32));
                mx = fmaxf(mx, __shfl_xor(mx, 8, 32));
                float newm = fmaxf(m[mt][r], mx);
                float corr = __expf(m[mt][r] - newm);
                p0[r] = __expf(v0 - newm);
                p1[r] = __expf(v1 - newm);
                float ps = p0[r] + p1[r];
                ps += __shfl_xor(ps, 1, 32);
                ps += __shfl_xor(ps, 2, 32);
                ps += __shfl_xor(ps, 4, 32);
                ps += __shfl_xor(ps, 8, 32);
                l[mt][r] = l[mt][r] * corr + ps;
                m[mt][r] = newm;
#pragma unroll
                for (int i = 0; i < 4; ++i) acc[mt][i][r] *= corr;
            }
            // Re-lay-out P (C layout) -> A-fragment layout through LDS.
#pragma unroll
            for (int r = 0; r < 8; ++r) {
                pbuf[mt][r + 8 * hi][col16]      = (__bf16)p0[r];
                pbuf[mt][r + 8 * hi][16 + col16] = (__bf16)p1[r];
            }
        }
        __syncthreads();
        Frag pf[2];
#pragma unroll
        for (int mt = 0; mt < 2; ++mt) {
            const __bf16* prow = &pbuf[mt][col16][0];
            pf[mt].q[0] = *(const uint4*)(prow + hi * 8);
            pf[mt].q[1] = *(const uint4*)(prow + 16 + hi * 8);
        }
        // V tile fragments, shared by both P row-tiles
#pragma unroll
        for (int i = 0; i < 4; ++i) {
            const __bf16* vrow = vT + (size_t)(h * HDIM + i * 16 + col16) * T_SEQ + s0;
            Frag bv;
            bv.q[0] = *(const uint4*)(vrow + hi * 8);
            bv.q[1] = *(const uint4*)(vrow + 16 + hi * 8);
            acc[0][i] = wmma_bf16(pf[0], bv, acc[0][i]);
            acc[1][i] = wmma_bf16(pf[1], bv, acc[1][i]);
        }
    }
#pragma unroll
    for (int mt = 0; mt < 2; ++mt)
#pragma unroll
        for (int i = 0; i < 4; ++i)
#pragma unroll
            for (int r = 0; r < 8; ++r) {
                size_t row = t0 + mt * 16 + r + 8 * hi;
                ob[row * DIM + h * HDIM + i * 16 + col16] =
                    (__bf16)(acc[mt][i][r] / l[mt][r]);
            }
}

// ---------------------------------------------------------------------------
// C = A(bf16 MxK, row-major) @ Bt(bf16 NxK, row-major == B^T) with epilogues.
// Block = 8 waves; each wave owns a 32x64 output tile (2x4 wmma accumulators,
// 8 wmma per k-step, B fragments reused by both A row-fragments).
// EPI 0: f32 store | 1: res + gate*v | 2: bf16 gelu(v+bias) | 3: res+gate*(v+bias)
// ---------------------------------------------------------------------------
template <int EPI>
__global__ __launch_bounds__(256) void gemm_bf16_kernel(
    const __bf16* __restrict__ A, const __bf16* __restrict__ Bt,
    int M, int N, int K, float* __restrict__ Cf, __bf16* __restrict__ Cb,
    const float* __restrict__ bias, const float* __restrict__ gate,
    const float* __restrict__ res) {
    const int tid = threadIdx.x;
    const int wave = tid >> 5, lane = tid & 31;
    const int col16 = lane & 15, hi = lane >> 4;
    const int m0 = blockIdx.x * 256 + wave * 32;
    const int n0 = blockIdx.y * 64;
    const __bf16* Arow0 = A + (size_t)(m0 + col16) * K;
    const __bf16* Arow1 = A + (size_t)(m0 + 16 + col16) * K;
    const __bf16* Brow  = Bt + (size_t)(n0 + col16) * K;
    v8f acc[2][4] = {};
    for (int k0 = 0; k0 < K; k0 += 32) {
        if (k0 + 32 < K) {
            __builtin_prefetch(Arow0 + k0 + 32, 0, 3);
            __builtin_prefetch(Arow1 + k0 + 32, 0, 3);
            __builtin_prefetch(Brow + k0 + 32, 0, 3);
        }
        Frag a0, a1;
        a0.q[0] = *(const uint4*)(Arow0 + k0 + hi * 8);
        a0.q[1] = *(const uint4*)(Arow0 + k0 + 16 + hi * 8);
        a1.q[0] = *(const uint4*)(Arow1 + k0 + hi * 8);
        a1.q[1] = *(const uint4*)(Arow1 + k0 + 16 + hi * 8);
#pragma unroll
        for (int i = 0; i < 4; ++i) {
            const __bf16* br = Brow + (size_t)i * 16 * K;
            Frag b;
            b.q[0] = *(const uint4*)(br + k0 + hi * 8);
            b.q[1] = *(const uint4*)(br + k0 + 16 + hi * 8);
            acc[0][i] = wmma_bf16(a0, b, acc[0][i]);
            acc[1][i] = wmma_bf16(a1, b, acc[1][i]);
        }
    }
#pragma unroll
    for (int mt = 0; mt < 2; ++mt) {
#pragma unroll
        for (int i = 0; i < 4; ++i) {
#pragma unroll
            for (int r = 0; r < 8; ++r) {
                size_t row = m0 + mt * 16 + r + 8 * hi;
                size_t col = n0 + i * 16 + col16;
                float v = acc[mt][i][r];
                if constexpr (EPI == 0) {
                    Cf[row * N + col] = v;
                } else if constexpr (EPI == 1) {
                    Cf[row * N + col] = res[row * N + col] + gate[col] * v;
                } else if constexpr (EPI == 2) {
                    Cb[row * N + col] = (__bf16)gelu_tanh(v + bias[col]);
                } else {
                    Cf[row * N + col] = res[row * N + col] + gate[col] * (v + bias[col]);
                }
            }
        }
    }
}

// ---------------------------------------------------------------------------
// Workspace layout
// ---------------------------------------------------------------------------
static constexpr size_t wal(size_t x) { return (x + 4095) & ~(size_t)4095; }
static constexpr size_t O_ADA   = 0;
static constexpr size_t O_HBF   = wal(O_ADA   + (size_t)ADA6 * 4);
static constexpr size_t O_WQKVT = wal(O_HBF   + (size_t)T_SEQ * DIM * 2);
static constexpr size_t O_QKV   = wal(O_WQKVT + (size_t)3 * DIM * DIM * 2);
static constexpr size_t O_QBF   = wal(O_QKV   + (size_t)T_SEQ * 3 * DIM * 4);
static constexpr size_t O_KBF   = wal(O_QBF   + (size_t)T_SEQ * DIM * 2);
static constexpr size_t O_VT    = wal(O_KBF   + (size_t)T_SEQ * DIM * 2);
static constexpr size_t O_WOUTT = wal(O_VT    + (size_t)T_SEQ * DIM * 2);
static constexpr size_t O_OBF   = wal(O_WOUTT + (size_t)DIM * DIM * 2);
static constexpr size_t O_X1    = wal(O_OBF   + (size_t)T_SEQ * DIM * 2);
static constexpr size_t O_H2BF  = wal(O_X1    + (size_t)T_SEQ * DIM * 4);
static constexpr size_t O_WM1T  = wal(O_H2BF  + (size_t)T_SEQ * DIM * 2);
static constexpr size_t O_M1BF  = wal(O_WM1T  + (size_t)MLPD * DIM * 2);
static constexpr size_t O_WM2T  = wal(O_M1BF  + (size_t)T_SEQ * MLPD * 2);

extern "C" void kernel_launch(void* const* d_in, const int* in_sizes, int n_in,
                              void* d_out, int out_size, void* d_ws, size_t ws_size,
                              hipStream_t stream) {
    const float* x      = (const float*)d_in[0];
    const float* c      = (const float*)d_in[1];
    const float* cosp   = (const float*)d_in[2];
    const float* sinp   = (const float*)d_in[3];
    const int*   seq    = (const int*)d_in[4];
    const float* ln1_w  = (const float*)d_in[5];
    const float* ln1_b  = (const float*)d_in[6];
    const float* w_qkv  = (const float*)d_in[7];
    const float* w_out  = (const float*)d_in[8];
    const float* ln2_w  = (const float*)d_in[9];
    const float* ln2_b  = (const float*)d_in[10];
    const float* w_mlp1 = (const float*)d_in[11];
    const float* b_mlp1 = (const float*)d_in[12];
    const float* w_mlp2 = (const float*)d_in[13];
    const float* b_mlp2 = (const float*)d_in[14];
    const float* w_ada  = (const float*)d_in[15];
    const float* b_ada  = (const float*)d_in[16];

    char* ws = (char*)d_ws;
    float*  ada   = (float*)(ws + O_ADA);
    __bf16* hbf   = (__bf16*)(ws + O_HBF);
    __bf16* wqkvT = (__bf16*)(ws + O_WQKVT);
    float*  qkv   = (float*)(ws + O_QKV);
    __bf16* qbf   = (__bf16*)(ws + O_QBF);
    __bf16* kbf   = (__bf16*)(ws + O_KBF);
    __bf16* vT    = (__bf16*)(ws + O_VT);
    __bf16* woutT = (__bf16*)(ws + O_WOUTT);
    __bf16* obf   = (__bf16*)(ws + O_OBF);
    float*  x1    = (float*)(ws + O_X1);
    __bf16* h2bf  = (__bf16*)(ws + O_H2BF);
    __bf16* wm1T  = (__bf16*)(ws + O_WM1T);
    __bf16* m1bf  = (__bf16*)(ws + O_M1BF);
    __bf16* wm2T  = (__bf16*)(ws + O_WM2T);

    const float* sh_msa = ada + 0 * DIM;
    const float* sc_msa = ada + 1 * DIM;
    const float* g_msa  = ada + 2 * DIM;
    const float* sh_mlp = ada + 3 * DIM;
    const float* sc_mlp = ada + 4 * DIM;
    const float* g_mlp  = ada + 5 * DIM;

    // 1) adaLN vector
    ada_kernel<<<ADA6 / 256, 256, 0, stream>>>(c, w_ada, b_ada, ada);

    // 2) weight conversions (f32 KxN -> bf16 NxK)
    convT_kernel<<<(DIM * 3 * DIM) / 256, 256, 0, stream>>>(w_qkv, wqkvT, DIM, 3 * DIM);
    convT_kernel<<<(DIM * DIM) / 256, 256, 0, stream>>>(w_out, woutT, DIM, DIM);
    convT_kernel<<<(DIM * MLPD) / 256, 256, 0, stream>>>(w_mlp1, wm1T, DIM, MLPD);
    convT_kernel<<<(MLPD * DIM) / 256, 256, 0, stream>>>(w_mlp2, wm2T, MLPD, DIM);

    // 3) LN1 + modulation -> bf16
    lnmod_kernel<<<T_SEQ, 256, 0, stream>>>(x, ln1_w, ln1_b, sc_msa, sh_msa, hbf);

    // 4) qkv = h @ w_qkv (f32 out)
    gemm_bf16_kernel<0><<<dim3(T_SEQ / 256, 3 * DIM / 64), 256, 0, stream>>>(
        hbf, wqkvT, T_SEQ, 3 * DIM, DIM, qkv, nullptr, nullptr, nullptr, nullptr);

    // 5) RoPE + split (V transposed)
    rope_split_kernel<<<T_SEQ, 512, 0, stream>>>(qkv, cosp, sinp, qbf, kbf, vT);

    // 6) masked flash attention -> o (bf16)
    attn_kernel<<<dim3(T_SEQ / 32, HEADS), 32, 0, stream>>>(qbf, kbf, vT, seq, obf);

    // 7) x1 = x + g_msa * (o @ w_out)
    gemm_bf16_kernel<1><<<dim3(T_SEQ / 256, DIM / 64), 256, 0, stream>>>(
        obf, woutT, T_SEQ, DIM, DIM, x1, nullptr, nullptr, g_msa, x);

    // 8) LN2 + modulation -> bf16
    lnmod_kernel<<<T_SEQ, 256, 0, stream>>>(x1, ln2_w, ln2_b, sc_mlp, sh_mlp, h2bf);

    // 9) m1 = gelu(h2 @ w_mlp1 + b_mlp1) -> bf16
    gemm_bf16_kernel<2><<<dim3(T_SEQ / 256, MLPD / 64), 256, 0, stream>>>(
        h2bf, wm1T, T_SEQ, MLPD, DIM, nullptr, m1bf, b_mlp1, nullptr, nullptr);

    // 10) out = x1 + g_mlp * (m1 @ w_mlp2 + b_mlp2)
    gemm_bf16_kernel<3><<<dim3(T_SEQ / 256, DIM / 64), 256, 0, stream>>>(
        m1bf, wm2T, T_SEQ, DIM, MLPD, (float*)d_out, nullptr, b_mlp2, g_mlp, x1);

    (void)in_sizes; (void)n_in; (void)out_size; (void)ws_size;
}
